// LSTM_80642305949766
// MI455X (gfx1250) — matrix operand order repaired and checked by reading, not compiled, not run on hardware
//
#include <hip/hip_runtime.h>
#include <hip/hip_bf16.h>

// ---------------- problem constants ----------------
#define TT 1024   // timesteps
#define BB 32     // batch
#define CC 256    // channels (input width for both layers, = 2H)
#define HH 128    // hidden per direction
#define GG 512    // 4H gate width
#define MM (TT * BB)  // 32768 GEMM rows

typedef __attribute__((ext_vector_type(16))) _Float16 v16h;
typedef __attribute__((ext_vector_type(8)))  _Float16 v8h;
typedef __attribute__((ext_vector_type(4)))  _Float16 v4h;
typedef __attribute__((ext_vector_type(8)))  float    v8f;

// ---------------- WMMA fragment helpers (contiguous k order k = 16*(lane/16) + i, same for A and B) ----------------

// A fragment: 16x32 f16, row-major source with leading dim `ld` (elements).
// lane<16: row=lane, K in {0..15}; lane>=16: row=lane-16, K in {16..31}
__device__ __forceinline__ v16h load_frag_a(const _Float16* base, int ld) {
  const int lane = threadIdx.x & 31;
  const int row  = lane & 15;
  const int koff = (lane >> 4) << 4;          // 0 or 16
  const _Float16* q = base + row * ld + koff;
  v8h lo = *reinterpret_cast<const v8h*>(q);       // K = koff .. koff+7
  v8h hi = *reinterpret_cast<const v8h*>(q + 8);   // K = koff+8 .. koff+15
  v16h a;
#pragma unroll
  for (int i = 0; i < 8; ++i) { a[i] = lo[i]; a[i + 8] = hi[i]; }
  return a;
}

// B fragment: 32x16 (KxN) f16 where B[k][n] = W[n0+n][k0+k], W row-major [N][K], ld = K stride.
// lane holds N = lane&15; K = 16*(lane>=16) + i, i=0..15 -> one contiguous 32B read of a W row.
__device__ __forceinline__ v16h load_frag_b(const _Float16* wbase, int ld) {
  const int lane = threadIdx.x & 31;
  const int n    = lane & 15;
  const int koff = (lane >> 4) << 4;          // 0 or 16
  return *reinterpret_cast<const v16h*>(wbase + n * ld + koff);
}

__device__ __forceinline__ v8f wmma_f16(v16h a, v16h b, v8f c) {
  return __builtin_amdgcn_wmma_f32_16x16x32_f16(false, a, false, b, (short)0, c, false, false);
}

__device__ __forceinline__ float sigf(float x) { return 1.0f / (1.0f + __expf(-x)); }
__device__ __forceinline__ float tanh_fast(float x) { return 2.0f * sigf(2.0f * x) - 1.0f; }

// ---------------- prep kernels ----------------

// f16 copies of weights + fused bias (b_ih + b_hh)
__global__ void prep_wb_kernel(const float* __restrict__ Wih, const float* __restrict__ Whh,
                               const float* __restrict__ bih, const float* __restrict__ bhh,
                               _Float16* __restrict__ WihH, _Float16* __restrict__ WhhH,
                               float* __restrict__ bsum) {
  const int NW_IH = 2 * 2 * GG * CC;   // 524288
  const int NW_HH = 2 * 2 * GG * HH;   // 262144
  const int NB    = 2 * 2 * GG;        // 2048
  int i = blockIdx.x * blockDim.x + threadIdx.x;
  if (i < NW_IH) {
    WihH[i] = (_Float16)Wih[i];
  } else if (i < NW_IH + NW_HH) {
    int k = i - NW_IH;
    WhhH[k] = (_Float16)Whh[k];
  } else if (i < NW_IH + NW_HH + NB) {
    int k = i - NW_IH - NW_HH;
    bsum[k] = bih[k] + bhh[k];
  }
}

// x [B][C][T] f32  ->  xh [T][B][C] f16 (time-major, coalesced reads over t)
__global__ void prep_x_kernel(const float* __restrict__ x, _Float16* __restrict__ xh) {
  size_t i = (size_t)blockIdx.x * blockDim.x + threadIdx.x;   // < B*C*T
  int t = (int)(i & (TT - 1));
  int tmp = (int)(i >> 10);
  int k = tmp & (CC - 1);
  int b = tmp >> 8;
  float v = x[((size_t)b * CC + k) * TT + t];
  xh[((size_t)t * BB + b) * CC + k] = (_Float16)v;
}

// hstage [T][B][C] f32 -> out [B][C][T] f32 (coalesced writes over t)
__global__ void finalize_out_kernel(const float* __restrict__ hstage, float* __restrict__ out) {
  size_t i = (size_t)blockIdx.x * blockDim.x + threadIdx.x;   // < B*C*T, i = (b*C + c)*T + t
  int t = (int)(i & (TT - 1));
  int tmp = (int)(i >> 10);
  int c = tmp & (CC - 1);
  int b = tmp >> 8;
  out[i] = hstage[((size_t)t * BB + b) * CC + c];
}

// ---------------- input-gate GEMM: gx[d] = X @ Wih[d]^T + (b_ih + b_hh) ----------------
// X: [MM][CC] f16, WihL: [2][GG][CC] f16.
// gx stored GATE-INTERLEAVED: [2][T*B][HH][4] f16, so the recurrence reads one
// contiguous v4h {i,f,g,o} per element.
// block 256 = 8 waves, each wave -> 16x64 output tile; grid (MM/128, GG/64, 2)
__global__ __launch_bounds__(256)
void gemm_gates_kernel(const _Float16* __restrict__ X,
                       const _Float16* __restrict__ WihL,
                       const float* __restrict__ bsumL,
                       _Float16* __restrict__ gx) {
  const int tid  = threadIdx.x;
  const int lane = tid & 31;
  const int wid  = tid >> 5;
  const int dir  = blockIdx.z;
  const size_t m0 = (size_t)blockIdx.x * 128 + (size_t)wid * 16;
  const int    n0 = blockIdx.y * 64;

  const _Float16* W   = WihL  + (size_t)dir * GG * CC;
  const float*    bd  = bsumL + dir * GG;
  _Float16*       gxd = gx    + (size_t)dir * MM * GG;

  v8f acc[4];
#pragma unroll
  for (int j = 0; j < 4; ++j) acc[j] = {};

#pragma unroll
  for (int kt = 0; kt < CC / 32; ++kt) {          // 8 K-tiles
    v16h a = load_frag_a(X + m0 * CC + kt * 32, CC);
#pragma unroll
    for (int j = 0; j < 4; ++j) {
      v16h b = load_frag_b(W + (size_t)(n0 + j * 16) * CC + kt * 32, CC);
      acc[j] = wmma_f16(a, b, acc[j]);
    }
  }

  const int hi   = (lane >> 4) & 1;
  const int nn   = lane & 15;
  const int gate = n0 >> 7;                       // uniform per block (n0 multiple of 64)
#pragma unroll
  for (int j = 0; j < 4; ++j)
#pragma unroll
    for (int r = 0; r < 8; ++r) {
      const int n_abs = n0 + j * 16 + nn;         // global gate-column
      const int jj    = n_abs & (HH - 1);         // column within gate
      // interleaved: [row][jj][gate]
      gxd[((m0 + r + hi * 8) * HH + jj) * 4 + gate] = (_Float16)(acc[j][r] + bd[n_abs]);
    }
}

// ---------------- persistent recurrence kernel ----------------
// grid.x = 2 (direction), block = 256 threads = 8 waves (2 waves/SIMD -> big VGPR budget).
// Wave `wid` owns column slice j0 = wid*16 of ALL FOUR gates, for all 32 batch rows:
//   * W_hh fragments (16 x v16h = 128 VGPRs) pinned in registers for the whole loop
//   * accumulators hold i/f/g/o for the wave's (b, j) elements -> cell update fully
//     in registers, cell state c in registers (16 floats/lane)
//   * gx pre-activations arrive as one v4h {i,f,g,o} load per element (16/step/thread)
//   * only h is shared: 2 x 8KB double-buffered LDS, ONE barrier per step
__global__ __launch_bounds__(256, 1)
void lstm_recur_kernel(const _Float16* __restrict__ gx,     // [2][T*B][HH][4], bias folded in
                       const _Float16* __restrict__ WhhL,   // [2][GG][HH]
                       _Float16* __restrict__ h_next_f16,   // [T][B][CC] f16 (layer 0 output)
                       float* __restrict__ hstage,          // [T][B][CC] f32 (last layer staging)
                       int is_last) {
  __shared__ __align__(32) _Float16 hS[2][BB][HH];   // 16 KB double-buffered hidden state

  const int dir  = blockIdx.x;
  const int tid  = threadIdx.x;
  const int lane = tid & 31;
  const int wid  = tid >> 5;            // 0..7
  const int j0   = wid * 16;            // column slice within each gate
  const int hi   = (lane >> 4) & 1;
  const int nn   = lane & 15;

  for (int e = tid; e < BB * HH; e += 256)
    hS[0][e >> 7][e & 127] = (_Float16)0.0f;

  const _Float16* Wd   = WhhL + (size_t)dir * GG * HH;
  const v4h*      gxd4 = reinterpret_cast<const v4h*>(gx) + (size_t)dir * (size_t)TT * BB * HH;

  // pin all recurrent-weight fragments in VGPRs (loop-invariant)
  v16h bfrag[4][4];                     // [gate][kt]
#pragma unroll
  for (int g = 0; g < 4; ++g)
#pragma unroll
    for (int kt = 0; kt < 4; ++kt)
      bfrag[g][kt] = load_frag_b(Wd + (size_t)(g * HH + j0) * HH + kt * 32, HH);

  float c_reg[2][8];                    // cell state, element (b = mt*16 + r + hi*8, j = j0+nn)
#pragma unroll
  for (int mt = 0; mt < 2; ++mt)
#pragma unroll
    for (int r = 0; r < 8; ++r) c_reg[mt][r] = 0.0f;

  __syncthreads();

  for (int s = 0; s < TT; ++s) {
    const int t = dir ? (TT - 1 - s) : s;
    const int p = s & 1;

    v8f acc[2][4];                      // [batch tile][gate]
#pragma unroll
    for (int mt = 0; mt < 2; ++mt)
#pragma unroll
      for (int g = 0; g < 4; ++g) acc[mt][g] = {};

#pragma unroll
    for (int mt = 0; mt < 2; ++mt)
#pragma unroll
      for (int kt = 0; kt < 4; ++kt) {
        v16h a = load_frag_a(&hS[p][mt * 16][0] + kt * 32, HH);
#pragma unroll
        for (int g = 0; g < 4; ++g)
          acc[mt][g] = wmma_f16(a, bfrag[g][kt], acc[mt][g]);
      }

    // in-register cell update (PyTorch gate order i, f, g, o)
#pragma unroll
    for (int mt = 0; mt < 2; ++mt)
#pragma unroll
      for (int r = 0; r < 8; ++r) {
        const int b = mt * 16 + r + hi * 8;
        const int j = j0 + nn;
        v4h gv = gxd4[((size_t)t * BB + b) * HH + j];   // {i,f,g,o} in one 8B load
        float gi = sigf(acc[mt][0][r] + (float)gv[0]);
        float gf = sigf(acc[mt][1][r] + (float)gv[1]);
        float gg = tanh_fast(acc[mt][2][r] + (float)gv[2]);
        float go = sigf(acc[mt][3][r] + (float)gv[3]);
        float cn = gf * c_reg[mt][r] + gi * gg;
        c_reg[mt][r] = cn;
        float h = go * tanh_fast(cn);
        hS[p ^ 1][b][j] = (_Float16)h;
        if (is_last)
          hstage[((size_t)t * BB + b) * CC + (dir * HH + j)] = h;            // coalesced f32
        else
          h_next_f16[((size_t)t * BB + b) * CC + (dir * HH + j)] = (_Float16)h;
      }
    __syncthreads();                    // single barrier per step (double-buffered h)
  }
}

// ---------------- launch ----------------
extern "C" void kernel_launch(void* const* d_in, const int* in_sizes, int n_in,
                              void* d_out, int out_size, void* d_ws, size_t ws_size,
                              hipStream_t stream) {
  const float* x   = (const float*)d_in[0];
  const float* Wih = (const float*)d_in[1];
  const float* Whh = (const float*)d_in[2];
  const float* bih = (const float*)d_in[3];
  const float* bhh = (const float*)d_in[4];
  float* out = (float*)d_out;

  char* ws = (char*)d_ws;
  // workspace layout (bytes), ~98.1 MB total
  _Float16* xh     = (_Float16*)(ws);                                   // 16 MB  [T][B][C] f16
  _Float16* h0h    = (_Float16*)(ws + ((size_t)16 << 20));              // 16 MB  [T][B][C] f16
  float*    hstage = (float*)   (ws);                                   // 32 MB  [T][B][C] f32, aliases xh+h0h
                                                                        //        (both dead after layer-1 GEMM)
  _Float16* gx     = (_Float16*)(ws + ((size_t)32 << 20));              // 64 MB  [2][T*B][HH][4] f16 (reused per layer)
  _Float16* WihH   = (_Float16*)(ws + ((size_t)96 << 20));              // 1 MB   [L][D][GG][CC] f16
  _Float16* WhhH   = (_Float16*)(ws + ((size_t)96 << 20) + (1u << 20)); // 512 KB [L][D][GG][HH] f16
  float*    bsum   = (float*)   (ws + ((size_t)96 << 20) + (2u << 20)); // 8 KB   [L][D][GG] f32

  // 1) weight/bias conversion + fused bias
  {
    int total = 2 * 2 * GG * CC + 2 * 2 * GG * HH + 2 * 2 * GG;         // 788480
    prep_wb_kernel<<<(total + 255) / 256, 256, 0, stream>>>(Wih, Whh, bih, bhh, WihH, WhhH, bsum);
  }
  // 2) x -> time-major f16
  prep_x_kernel<<<(BB * CC * TT) / 256, 256, 0, stream>>>(x, xh);

  dim3 ggrid(MM / 128, GG / 64, 2);

  // layer 0
  gemm_gates_kernel<<<ggrid, 256, 0, stream>>>(xh, WihH, bsum, gx);
  lstm_recur_kernel<<<2, 256, 0, stream>>>(gx, WhhH, h0h, hstage, 0);

  // layer 1 (input = concat(fwd,bwd) = h0h; gx buffer reused — ordering via stream)
  gemm_gates_kernel<<<ggrid, 256, 0, stream>>>(h0h, WihH + 2 * GG * CC, bsum + 2 * GG, gx);
  lstm_recur_kernel<<<2, 256, 0, stream>>>(gx, WhhH + 2 * GG * HH, h0h /*unused*/, hstage, 1);

  // final transpose [T][B][C] -> [B][C][T]
  finalize_out_kernel<<<(BB * CC * TT) / 256, 256, 0, stream>>>(hstage, out);
}
